// PHPointNet_89541478187053
// MI455X (gfx1250) — compile-verified, matrix operand order checked by reading, not run on hardware
//
#include <hip/hip_runtime.h>
#include <hip/hip_bf16.h>

#define NN   100000
#define NE   3200000
#define NG   16
#define HD   32
#define PHD  64
#define NC   10
#define BN_EPS 1e-5f

// edge-tile geometry: NE/16 = 200000 tiles; 2000 blocks * 4 waves = 8000 waves
// -> exactly 25 tiles per wave (uniform trip count, block barriers stay legal)
#define WAVES      4
#define EDGE_BLKS  2000
#define TOT_WAVES  (EDGE_BLKS * WAVES)
#define NTILES     (NE / 16)

typedef __attribute__((ext_vector_type(16))) _Float16 v16h;
typedef __attribute__((ext_vector_type(8)))  float    v8f;

// ---- order-preserving float <-> uint key for atomic max ----
__device__ __forceinline__ unsigned fkey(float f) {
  unsigned u = __float_as_uint(f);
  return (u & 0x80000000u) ? ~u : (u | 0x80000000u);
}
__device__ __forceinline__ float fdec(unsigned k) {
  unsigned u = (k & 0x80000000u) ? (k & 0x7FFFFFFFu) : ~k;  // key 0 -> NaN
  return __uint_as_float(u);
}
__device__ __forceinline__ int finitef_(float v) {
  return (__float_as_uint(v) & 0x7F800000u) != 0x7F800000u;
}
// A/B fragment element -> K index (16-bit 16x32 WMMA layout, per lane-half)
__device__ __forceinline__ int kmap(int e, int hf) {
  return (e < 8) ? (hf * 8 + e) : (16 + hf * 8 + (e - 8));
}

// ---------------- init: zero agg keys + BN sums ----------------
__global__ void k_init(unsigned* p, int n) {
  int i = blockIdx.x * blockDim.x + threadIdx.x;
  if (i < n) p[i] = 0u;
}

// ---------------- layer1 BN stats: x = feat6 @ W1 + b1 ----------------
__global__ void k_stats1(const float* __restrict__ pos,
                         const int* __restrict__ src, const int* __restrict__ dst,
                         const float* __restrict__ W1, const float* __restrict__ b1,
                         float* __restrict__ sums) {
  __shared__ float ls[64];
  int t = threadIdx.x;
  if (t < 64) ls[t] = 0.f;
  __syncthreads();
  int e = blockIdx.x * blockDim.x + t;
  int s = src[e], d = dst[e];
  float f0 = pos[3*s], f1 = pos[3*s+1], f2 = pos[3*s+2];
  float f3 = f0 - pos[3*d], f4 = f1 - pos[3*d+1], f5 = f2 - pos[3*d+2];
#pragma unroll
  for (int n = 0; n < 32; ++n) {
    float x = b1[n] + f0*W1[n] + f1*W1[32+n] + f2*W1[64+n]
                    + f3*W1[96+n] + f4*W1[128+n] + f5*W1[160+n];
    atomicAdd(&ls[n], x);
    atomicAdd(&ls[32+n], x*x);
  }
  __syncthreads();
  if (t < 64) atomicAdd(&sums[t], ls[t]);
}

// ---------------- layer2 BN stats: x = [h32|dpos3] @ W1b + b1b ----------------
__global__ void k_stats2(const float* __restrict__ pos, const float* __restrict__ h1,
                         const int* __restrict__ src, const int* __restrict__ dst,
                         const float* __restrict__ W1, const float* __restrict__ b1,
                         float* __restrict__ sums) {
  __shared__ float ls[64];
  int t = threadIdx.x;
  if (t < 64) ls[t] = 0.f;
  __syncthreads();
  int e = blockIdx.x * blockDim.x + t;
  int s = src[e], d = dst[e];
  float hs[32];
#pragma unroll
  for (int k = 0; k < 32; ++k) hs[k] = h1[s*32 + k];
  float d0 = pos[3*s]   - pos[3*d];
  float d1 = pos[3*s+1] - pos[3*d+1];
  float d2 = pos[3*s+2] - pos[3*d+2];
#pragma unroll 4
  for (int n = 0; n < 32; ++n) {
    float x = b1[n];
#pragma unroll
    for (int k = 0; k < 32; ++k) x += hs[k] * W1[k*32 + n];
    x += d0*W1[32*32+n] + d1*W1[33*32+n] + d2*W1[34*32+n];
    atomicAdd(&ls[n], x);
    atomicAdd(&ls[32+n], x*x);
  }
  __syncthreads();
  if (t < 64) atomicAdd(&sums[t], ls[t]);
}

// ---------------- layer1 main pass (WMMA): BN+relu+GEMM2+max-scatter ----------------
__global__ __launch_bounds__(128) void k_edges1(
    const float* __restrict__ pos, const int* __restrict__ src, const int* __restrict__ dst,
    const float* __restrict__ W1, const float* __restrict__ b1,
    const float* __restrict__ gamma, const float* __restrict__ beta,
    const float* __restrict__ W2, const float* __restrict__ b2,
    const float* __restrict__ sums, unsigned* __restrict__ agg) {
  __shared__ _Float16 tile[WAVES][16][40];
  const int lane = threadIdx.x & 31;
  const int w    = threadIdx.x >> 5;
  const int wgid = blockIdx.x * WAVES + w;   // global wave id in [0, TOT_WAVES)
  const int n  = lane & 15;
  const int hf = lane >> 4;
  __builtin_prefetch(W2, 0, 0);

  // ---- loop-invariant: B fragments (W1 K=6 padded to 32; W2 K=32), 2 N-blocks ----
  v16h bw1_0, bw1_1, bw2_0, bw2_1;
#pragma unroll
  for (int e = 0; e < 16; ++e) {
    int k = kmap(e, hf);
    bw1_0[e] = (_Float16)((k < 6) ? W1[k*32 + n]      : 0.f);
    bw1_1[e] = (_Float16)((k < 6) ? W1[k*32 + 16 + n] : 0.f);
    bw2_0[e] = (_Float16)W2[k*32 + n];
    bw2_1[e] = (_Float16)W2[k*32 + 16 + n];
  }
  // ---- loop-invariant: BN constants (stats x already included b1 -> fold) ----
  const float invE = 1.f / (float)NE;
  float m0 = sums[n]    * invE, m1 = sums[16+n]    * invE;
  float v0 = sums[32+n] * invE - m0*m0, v1 = sums[48+n] * invE - m1*m1;
  float sc0 = gamma[n]    * rsqrtf(v0 + BN_EPS);
  float sc1 = gamma[16+n] * rsqrtf(v1 + BN_EPS);
  float sh0 = (b1[n]    - m0) * sc0 + beta[n];
  float sh1 = (b1[16+n] - m1) * sc1 + beta[16+n];
  float bb0 = b2[n], bb1 = b2[16+n];

  // ---- uniform-trip-count tile loop (NTILES % TOT_WAVES == 0) ----
  for (int tileId = wgid; tileId < NTILES; tileId += TOT_WAVES) {
    const int base = tileId * 16;

    // stage 16x6 feat tile (padded to 32) in LDS as f16
    if (lane < 16) {
      int e = base + lane;
      int s = src[e], d = dst[e];
      float f0 = pos[3*s], f1 = pos[3*s+1], f2 = pos[3*s+2];
      _Float16* row = &tile[w][lane][0];
      row[0] = (_Float16)f0; row[1] = (_Float16)f1; row[2] = (_Float16)f2;
      row[3] = (_Float16)(f0 - pos[3*d]);
      row[4] = (_Float16)(f1 - pos[3*d+1]);
      row[5] = (_Float16)(f2 - pos[3*d+2]);
#pragma unroll
      for (int j = 6; j < 32; ++j) row[j] = (_Float16)0.f;
    }
    __syncthreads();

    v16h a;
#pragma unroll
    for (int e = 0; e < 16; ++e) a[e] = tile[w][n][kmap(e, hf)];

    v8f c0 = {}, c1 = {};
    c0 = __builtin_amdgcn_wmma_f32_16x16x32_f16(false, a, false, bw1_0, (short)0, c0, false, false);
    c1 = __builtin_amdgcn_wmma_f32_16x16x32_f16(false, a, false, bw1_1, (short)0, c1, false, false);
    __syncthreads();

    // BN + relu, write transposed back to LDS as A fragment source for GEMM2
#pragma unroll
    for (int r = 0; r < 8; ++r) {
      tile[w][r + 8*hf][n]      = (_Float16)fmaxf(c0[r]*sc0 + sh0, 0.f);
      tile[w][r + 8*hf][16 + n] = (_Float16)fmaxf(c1[r]*sc1 + sh1, 0.f);
    }
    __syncthreads();

    v16h a2;
#pragma unroll
    for (int e = 0; e < 16; ++e) a2[e] = tile[w][n][kmap(e, hf)];
    v8f y0 = {}, y1 = {};
    y0 = __builtin_amdgcn_wmma_f32_16x16x32_f16(false, a2, false, bw2_0, (short)0, y0, false, false);
    y1 = __builtin_amdgcn_wmma_f32_16x16x32_f16(false, a2, false, bw2_1, (short)0, y1, false, false);

#pragma unroll
    for (int r = 0; r < 8; ++r) {
      int e = base + r + 8*hf;
      unsigned dnode = (unsigned)dst[e];
      atomicMax(&agg[dnode*32 + n],      fkey(y0[r] + bb0));
      atomicMax(&agg[dnode*32 + 16 + n], fkey(y1[r] + bb1));
    }
    __syncthreads();   // keep next iteration's staging from racing fragment reads
  }
}

// ---------------- layer2 main pass (WMMA, K=35 -> two K-tiles) ----------------
__global__ __launch_bounds__(128) void k_edges2(
    const float* __restrict__ pos, const float* __restrict__ h1,
    const int* __restrict__ src, const int* __restrict__ dst,
    const float* __restrict__ W1, const float* __restrict__ b1,
    const float* __restrict__ gamma, const float* __restrict__ beta,
    const float* __restrict__ W2, const float* __restrict__ b2,
    const float* __restrict__ sums, unsigned* __restrict__ agg) {
  __shared__ _Float16 tile[WAVES][16][72];
  const int lane = threadIdx.x & 31;
  const int w    = threadIdx.x >> 5;
  const int wgid = blockIdx.x * WAVES + w;
  const int n  = lane & 15;
  const int hf = lane >> 4;
  __builtin_prefetch(W1, 0, 0);

  // ---- loop-invariant: B fragments (W1b 35->64 over two K-tiles; W2b K=32) ----
  v16h b00, b01, b10, b11, bw2_0, bw2_1;
#pragma unroll
  for (int e = 0; e < 16; ++e) {
    int k = kmap(e, hf);
    int k2 = 32 + k;
    b00[e] = (_Float16)W1[k*32 + n];
    b01[e] = (_Float16)W1[k*32 + 16 + n];
    b10[e] = (_Float16)((k2 < 35) ? W1[k2*32 + n]      : 0.f);
    b11[e] = (_Float16)((k2 < 35) ? W1[k2*32 + 16 + n] : 0.f);
    bw2_0[e] = (_Float16)W2[k*32 + n];
    bw2_1[e] = (_Float16)W2[k*32 + 16 + n];
  }
  const float invE = 1.f / (float)NE;
  float m0 = sums[n]    * invE, m1 = sums[16+n]    * invE;
  float v0 = sums[32+n] * invE - m0*m0, v1 = sums[48+n] * invE - m1*m1;
  float sc0 = gamma[n]    * rsqrtf(v0 + BN_EPS);
  float sc1 = gamma[16+n] * rsqrtf(v1 + BN_EPS);
  float sh0 = (b1[n]    - m0) * sc0 + beta[n];
  float sh1 = (b1[16+n] - m1) * sc1 + beta[16+n];
  float bb0 = b2[n], bb1 = b2[16+n];

  for (int tileId = wgid; tileId < NTILES; tileId += TOT_WAVES) {
    const int base = tileId * 16;

    if (lane < 16) {
      int e = base + lane;
      int s = src[e], d = dst[e];
      _Float16* row = &tile[w][lane][0];
#pragma unroll
      for (int j = 0; j < 32; ++j) row[j] = (_Float16)h1[s*32 + j];
      row[32] = (_Float16)(pos[3*s]   - pos[3*d]);
      row[33] = (_Float16)(pos[3*s+1] - pos[3*d+1]);
      row[34] = (_Float16)(pos[3*s+2] - pos[3*d+2]);
#pragma unroll
      for (int j = 35; j < 64; ++j) row[j] = (_Float16)0.f;
    }
    __syncthreads();

    v16h a0, a1;
#pragma unroll
    for (int e = 0; e < 16; ++e) {
      int k = kmap(e, hf);
      a0[e] = tile[w][n][k];
      a1[e] = tile[w][n][32 + k];
    }
    v8f c0 = {}, c1 = {};
    c0 = __builtin_amdgcn_wmma_f32_16x16x32_f16(false, a0, false, b00, (short)0, c0, false, false);
    c0 = __builtin_amdgcn_wmma_f32_16x16x32_f16(false, a1, false, b10, (short)0, c0, false, false);
    c1 = __builtin_amdgcn_wmma_f32_16x16x32_f16(false, a0, false, b01, (short)0, c1, false, false);
    c1 = __builtin_amdgcn_wmma_f32_16x16x32_f16(false, a1, false, b11, (short)0, c1, false, false);
    __syncthreads();

#pragma unroll
    for (int r = 0; r < 8; ++r) {
      tile[w][r + 8*hf][n]      = (_Float16)fmaxf(c0[r]*sc0 + sh0, 0.f);
      tile[w][r + 8*hf][16 + n] = (_Float16)fmaxf(c1[r]*sc1 + sh1, 0.f);
    }
    __syncthreads();

    v16h a2;
#pragma unroll
    for (int e = 0; e < 16; ++e) a2[e] = tile[w][n][kmap(e, hf)];
    v8f y0 = {}, y1 = {};
    y0 = __builtin_amdgcn_wmma_f32_16x16x32_f16(false, a2, false, bw2_0, (short)0, y0, false, false);
    y1 = __builtin_amdgcn_wmma_f32_16x16x32_f16(false, a2, false, bw2_1, (short)0, y1, false, false);

#pragma unroll
    for (int r = 0; r < 8; ++r) {
      int e = base + r + 8*hf;
      unsigned dnode = (unsigned)dst[e];
      atomicMax(&agg[dnode*32 + n],      fkey(y0[r] + bb0));
      atomicMax(&agg[dnode*32 + 16 + n], fkey(y1[r] + bb1));
    }
    __syncthreads();
  }
}

// ---------------- finish layer1: h1 = relu(finite(agg1)) ----------------
__global__ void k_finish1(const unsigned* __restrict__ agg, float* __restrict__ h1) {
  int i = blockIdx.x * blockDim.x + threadIdx.x;
  if (i >= NN * 32) return;
  float v = fdec(agg[i]);
  h1[i] = finitef_(v) ? fmaxf(v, 0.f) : 0.f;
}

// ---------------- finish layer2 + graph pooling ----------------
__global__ void k_finish2(const unsigned* __restrict__ agg, const int* __restrict__ batch,
                          unsigned* __restrict__ gagg) {
  int i = blockIdx.x * blockDim.x + threadIdx.x;
  if (i >= NN * 32) return;
  int node = i >> 5, c = i & 31;
  float v = fdec(agg[i]);
  float h = finitef_(v) ? fmaxf(v, 0.f) : 0.f;
  atomicMax(&gagg[(unsigned)batch[node]*32 + c], fkey(h));
}

// ---------------- classifier head ----------------
__global__ void k_head(const unsigned* __restrict__ gagg, const float* __restrict__ ph,
                       const float* __restrict__ Wc, const float* __restrict__ bc,
                       float* __restrict__ out) {
  int i = blockIdx.x * blockDim.x + threadIdx.x;
  if (i >= NG * NC) return;
  int g = i / NC, j = i % NC;
  float acc = bc[j];
#pragma unroll
  for (int c = 0; c < 32; ++c) {
    float v = fdec(gagg[g*32 + c]);
    if (!finitef_(v)) v = 0.f;
    acc += v * Wc[c*NC + j];
  }
#pragma unroll 8
  for (int p = 0; p < PHD; ++p) acc += ph[g*PHD + p] * Wc[(32 + p)*NC + j];
  out[i] = acc;
}

extern "C" void kernel_launch(void* const* d_in, const int* in_sizes, int n_in,
                              void* d_out, int out_size, void* d_ws, size_t ws_size,
                              hipStream_t stream) {
  const float* pos  = (const float*)d_in[0];
  const float* ph   = (const float*)d_in[1];
  const int*   ei   = (const int*)d_in[2];
  const int*   batch= (const int*)d_in[3];
  const float* W1a  = (const float*)d_in[4];
  const float* b1a  = (const float*)d_in[5];
  const float* g1   = (const float*)d_in[6];
  const float* bt1  = (const float*)d_in[7];
  const float* W2a  = (const float*)d_in[8];
  const float* b2a  = (const float*)d_in[9];
  const float* W1b  = (const float*)d_in[10];
  const float* b1b  = (const float*)d_in[11];
  const float* g2   = (const float*)d_in[12];
  const float* bt2  = (const float*)d_in[13];
  const float* W2b  = (const float*)d_in[14];
  const float* b2b  = (const float*)d_in[15];
  const float* Wc   = (const float*)d_in[16];
  const float* bc   = (const float*)d_in[17];
  const int* src = ei;
  const int* dst = ei + NE;
  float* out = (float*)d_out;

  // workspace layout
  unsigned* agg1  = (unsigned*)d_ws;           // NN*32
  unsigned* agg2  = agg1 + (size_t)NN * 32;    // NN*32
  unsigned* gagg  = agg2 + (size_t)NN * 32;    // NG*32
  float*    sums1 = (float*)(gagg + NG * 32);  // 64
  float*    sums2 = sums1 + 64;                // 64
  float*    h1    = sums2 + 64;                // NN*32

  const int initWords = NN * 32 * 2 + NG * 32 + 128;
  k_init<<<(initWords + 255) / 256, 256, 0, stream>>>(agg1, initWords);

  // layer 1
  k_stats1<<<NE / 256, 256, 0, stream>>>(pos, src, dst, W1a, b1a, sums1);
  k_edges1<<<EDGE_BLKS, 128, 0, stream>>>(pos, src, dst, W1a, b1a, g1, bt1,
                                          W2a, b2a, sums1, agg1);
  k_finish1<<<(NN * 32 + 255) / 256, 256, 0, stream>>>(agg1, h1);

  // layer 2
  k_stats2<<<NE / 256, 256, 0, stream>>>(pos, h1, src, dst, W1b, b1b, sums2);
  k_edges2<<<EDGE_BLKS, 128, 0, stream>>>(pos, h1, src, dst, W1b, b1b, g2, bt2,
                                          W2b, b2b, sums2, agg2);
  k_finish2<<<(NN * 32 + 255) / 256, 256, 0, stream>>>(agg2, batch, gagg);

  // head
  k_head<<<1, 256, 0, stream>>>(gagg, ph, Wc, bc, out);
}
